// AstrocyteMemoryModule_42700564857156
// MI455X (gfx1250) — compile-verified, matrix operand order checked by reading, not run on hardware
//
#include <hip/hip_runtime.h>
#include <hip/hip_bf16.h>
#include <cstdint>

// Problem dims (fixed by the reference)
#define D_DIM  1024
#define M_DIM  8192
#define H_DIM  16
#define B_DIM  64
#define HD_DIM 64
#define AT_CHUNK 512
#define NC (M_DIM / AT_CHUNK)   // 16 attention M-chunks

typedef __attribute__((ext_vector_type(16))) __bf16 v16bf;
typedef __attribute__((ext_vector_type(8)))  float  v8f;

// ---------------------------------------------------------------------------
// CDNA5 async global->LDS copy (ASYNCcnt-tracked), one b128 per lane.
// LDS byte offset in vdst ; global addr = vaddr pair (GV mode, saddr=off).
// ---------------------------------------------------------------------------
__device__ __forceinline__ void async_copy_b128(uint32_t lds_off, const void* gptr) {
  asm volatile("global_load_async_to_lds_b128 %0, %1, off"
               :: "v"(lds_off), "v"((unsigned long long)(uintptr_t)gptr)
               : "memory");
}
__device__ __forceinline__ uint32_t lds_off32(const void* p) {
  // generic pointer to LDS: addr[31:0] is the LDS byte offset (ISA 10.2)
  return (uint32_t)(uintptr_t)p;
}

// ---------------------------------------------------------------------------
// Load one 16x32 bf16 A/B operand fragment from an LDS tile (row-major, ld
// bf16 elems per row) per the CDNA5 WMMA 16-bit operand layout:
//   lanes 0-15 : row = row0+lane,    K halves [0..8) then [16..24)
//   lanes16-31 : row = row0+lane-16, K halves [8..16) then [24..32)
// ---------------------------------------------------------------------------
__device__ __forceinline__ v16bf load_frag(const __bf16* tile, int ld, int row0, int lane) {
  int r = row0 + (lane & 15);
  int k = (lane & 16) ? 8 : 0;
  union { uint4 u[2]; v16bf v; } p;
  p.u[0] = *reinterpret_cast<const uint4*>(tile + r * ld + k);
  p.u[1] = *reinterpret_cast<const uint4*>(tile + r * ld + k + 16);
  return p.v;
}

__device__ __forceinline__ v8f vzero8() {
  v8f z = {0.f, 0.f, 0.f, 0.f, 0.f, 0.f, 0.f, 0.f};
  return z;
}

// ---------------------------------------------------------------------------
// GEMM: C[M x N] = act(A[M x K] * W[N x K]^T + bias[N])
//  - A: f32 (AF32, converted while staging) or bf16 (async b128 -> LDS)
//  - W: bf16 row-major (pre-converted), staged with async b128 -> LDS
//  - ACT: 0 none, 1 relu, 2 sigmoid; out bf16 (Cb) and/or f32 (Cf)
// 256 threads = 8 wave32 waves; C tile 128x128; K step 32; double-buffered
// LDS with ASYNCcnt-based pipelining. Requires N%128==0, K%32==0; M handled
// via row clamp + store guard.
// Wave w: rows [64*(w>>2), +64) (4 subtiles), cols [32*(w&3), +32) (2 subtiles).
// ---------------------------------------------------------------------------
template <int ACT, bool AF32>
__global__ __launch_bounds__(256) void gemm_bt_kernel(
    const void* __restrict__ Av, int lda,
    const __bf16* __restrict__ W, int ldw,
    const float* __restrict__ bias,
    __bf16* __restrict__ Cb, float* __restrict__ Cf, int ldc,
    int Mtot, int K)
{
  __shared__ __bf16 As[2][128 * 32];
  __shared__ __bf16 Bs[2][128 * 32];

  const int m0   = blockIdx.y * 128;
  const int n0   = blockIdx.x * 128;
  const int tid  = threadIdx.x;
  const int lane = tid & 31;
  const int wr   = (tid >> 5) >> 2;   // 0..1  (64-row half)
  const int wc   = (tid >> 5) & 3;    // 0..3  (32-col quarter)

  v8f acc[4][2];
#pragma unroll
  for (int i = 0; i < 4; ++i)
#pragma unroll
    for (int j = 0; j < 2; ++j) acc[i][j] = vzero8();

  // ---- tile staging ----
  auto stageA = [&](int buf, int k0) {
    if (AF32) {
      const float* A = (const float*)Av;
#pragma unroll
      for (int it = 0; it < 4; ++it) {
        int idx = tid + it * 256;             // 1024 quads of 4 floats
        int r = idx >> 3, c4 = (idx & 7) * 4;
        int rg = m0 + r; rg = (rg < Mtot) ? rg : (Mtot - 1);
        float4 f = *reinterpret_cast<const float4*>(A + (size_t)rg * lda + k0 + c4);
        union { __bf16 h[4]; uint64_t u; } p;
        p.h[0] = (__bf16)f.x; p.h[1] = (__bf16)f.y;
        p.h[2] = (__bf16)f.z; p.h[3] = (__bf16)f.w;
        *reinterpret_cast<uint64_t*>(&As[buf][r * 32 + c4]) = p.u;
      }
    } else {
      const __bf16* A = (const __bf16*)Av;
#pragma unroll
      for (int it = 0; it < 2; ++it) {
        int idx = tid + it * 256;             // 512 chunks of 8 halves
        int r = idx >> 2, c8 = (idx & 3) * 8;
        int rg = m0 + r; rg = (rg < Mtot) ? rg : (Mtot - 1);
        async_copy_b128(lds_off32(&As[buf][r * 32 + c8]),
                        A + (size_t)rg * lda + k0 + c8);
      }
    }
  };
  auto stageB = [&](int buf, int k0) {
#pragma unroll
    for (int it = 0; it < 2; ++it) {
      int idx = tid + it * 256;
      int r = idx >> 2, c8 = (idx & 3) * 8;
      async_copy_b128(lds_off32(&Bs[buf][r * 32 + c8]),
                      W + (size_t)(n0 + r) * ldw + k0 + c8);
    }
  };

  // async ops in flight per wave for one prefetched stage
  constexpr int PF = AF32 ? 2 : 4;   // B only (2) or A+B (4)

  stageA(0, 0);
  stageB(0, 0);

  const int T = K / 32;
  for (int t = 0; t < T; ++t) {
    const int cur = t & 1;
    const bool pf = (t + 1 < T);
    if (pf) { stageA(cur ^ 1, (t + 1) * 32); stageB(cur ^ 1, (t + 1) * 32); }
    if (pf) asm volatile("s_wait_asynccnt %0" :: "n"(PF) : "memory");
    else    asm volatile("s_wait_asynccnt 0" ::: "memory");
    __syncthreads();

    v16bf afr[4], bfr[2];
#pragma unroll
    for (int i = 0; i < 4; ++i) afr[i] = load_frag(&As[cur][0], 32, wr * 64 + i * 16, lane);
#pragma unroll
    for (int j = 0; j < 2; ++j) bfr[j] = load_frag(&Bs[cur][0], 32, wc * 32 + j * 16, lane);
#pragma unroll
    for (int j = 0; j < 2; ++j)
#pragma unroll
      for (int i = 0; i < 4; ++i)
        acc[i][j] = __builtin_amdgcn_wmma_f32_16x16x32_bf16(
            false, afr[i], false, bfr[j], (short)0, acc[i][j], false, false);
    __syncthreads();
  }

  // ---- epilogue: bias + activation + store ----
  const int rhi = (lane & 16) ? 8 : 0;
#pragma unroll
  for (int j = 0; j < 2; ++j) {
    const int col = n0 + wc * 32 + j * 16 + (lane & 15);
    const float bb = bias[col];
#pragma unroll
    for (int i = 0; i < 4; ++i) {
#pragma unroll
      for (int r = 0; r < 8; ++r) {
        int row = m0 + wr * 64 + i * 16 + r + rhi;
        if (row < Mtot) {
          float v = acc[i][j][r] + bb;
          if (ACT == 1) v = fmaxf(v, 0.f);
          if (ACT == 2) v = 1.f / (1.f + __expf(-v));
          if (Cb) Cb[(size_t)row * ldc + col] = (__bf16)v;
          if (Cf) Cf[(size_t)row * ldc + col] = v;
        }
      }
    }
  }
}

// ---------------------------------------------------------------------------
// Attention partial (split-M flash-decoding, no max subtraction: scores are
// O(1e-2) so plain exp is numerically safe). One workgroup per (head h,
// M-chunk c of 512). pnum[h][c][b][d] = sum exp(s)*v ; pden[h][c][b] = sum exp(s).
// ---------------------------------------------------------------------------
__global__ __launch_bounds__(128) void attn_partial_kernel(
    const __bf16* __restrict__ qh,   // [64 x 1024]
    const __bf16* __restrict__ kh,   // [8192 x 1024]
    const __bf16* __restrict__ vh,   // [8192 x 1024]
    float* __restrict__ pnum,        // [H][NC][64][64]
    float* __restrict__ pden)        // [H][NC][64]
{
  const int h    = blockIdx.x;
  const int c    = blockIdx.y;
  const int lane = threadIdx.x & 31;
  const int w    = threadIdx.x >> 5;

  __shared__ __bf16 Qs[64 * 64];
  __shared__ __bf16 Ks[64 * 64];
  __shared__ __bf16 Ps[64 * 64];
  __shared__ __bf16 Vt[64 * 64];   // V tile transposed: Vt[d][j]
  __shared__ float  den[64];

  {
    const __bf16* src = qh + h * 64;
#pragma unroll
    for (int it = 0; it < 4; ++it) {
      int idx = threadIdx.x + it * 128;
      int r = idx >> 3, c8 = (idx & 7) * 8;
      *reinterpret_cast<uint4*>(Qs + r * 64 + c8) =
          *reinterpret_cast<const uint4*>(src + (size_t)r * D_DIM + c8);
    }
  }
  if (threadIdx.x < 64) den[threadIdx.x] = 0.f;

  v8f acc[4];
#pragma unroll
  for (int i = 0; i < 4; ++i) acc[i] = vzero8();
  __syncthreads();

  for (int s = 0; s < AT_CHUNK / 64; ++s) {
    const int m64 = c * AT_CHUNK + s * 64;

    {
      const __bf16* src = kh + (size_t)m64 * D_DIM + h * 64;
#pragma unroll
      for (int it = 0; it < 4; ++it) {
        int idx = threadIdx.x + it * 128;
        int r = idx >> 3, c8 = (idx & 7) * 8;
        *reinterpret_cast<uint4*>(Ks + r * 64 + c8) =
            *reinterpret_cast<const uint4*>(src + (size_t)r * D_DIM + c8);
      }
    }
    __syncthreads();

    // S = Q @ K^T
    v8f sacc[4];
#pragma unroll
    for (int i = 0; i < 4; ++i) sacc[i] = vzero8();
#pragma unroll
    for (int kk = 0; kk < 64; kk += 32) {
      v16bf bf = load_frag(Ks + kk, 64, w * 16, lane);
#pragma unroll
      for (int i = 0; i < 4; ++i) {
        v16bf af = load_frag(Qs + kk, 64, i * 16, lane);
        sacc[i] = __builtin_amdgcn_wmma_f32_16x16x32_bf16(
            false, af, false, bf, (short)0, sacc[i], false, false);
      }
    }

    // P = exp(S/8); D->A re-layout via LDS; accumulate denominators
    {
      const int colp = w * 16 + (lane & 15);
      const int rhi  = (lane & 16) ? 8 : 0;
#pragma unroll
      for (int i = 0; i < 4; ++i) {
#pragma unroll
        for (int r = 0; r < 8; ++r) {
          int row = i * 16 + r + rhi;
          float e = __expf(sacc[i][r] * 0.125f);
          Ps[row * 64 + colp] = (__bf16)e;
          atomicAdd(&den[row], e);
        }
      }
    }

    // stage V tile transposed (V columns -> contiguous B-operand rows)
    {
      int j  = threadIdx.x >> 1;
      int d0 = (threadIdx.x & 1) * 32;
      const uint4* vp4 = reinterpret_cast<const uint4*>(
          vh + (size_t)(m64 + j) * D_DIM + h * 64 + d0);
      union { uint4 u[4]; __bf16 hb[32]; } buf;
      buf.u[0] = vp4[0]; buf.u[1] = vp4[1]; buf.u[2] = vp4[2]; buf.u[3] = vp4[3];
#pragma unroll
      for (int d = 0; d < 32; ++d) Vt[(d0 + d) * 64 + j] = buf.hb[d];
    }
    __syncthreads();

    // ctx += P @ V
#pragma unroll
    for (int kk = 0; kk < 64; kk += 32) {
      v16bf bf = load_frag(Vt + kk, 64, w * 16, lane);
#pragma unroll
      for (int i = 0; i < 4; ++i) {
        v16bf af = load_frag(Ps + kk, 64, i * 16, lane);
        acc[i] = __builtin_amdgcn_wmma_f32_16x16x32_bf16(
            false, af, false, bf, (short)0, acc[i], false, false);
      }
    }
    __syncthreads();
  }

  {
    const int colp = w * 16 + (lane & 15);
    const int rhi  = (lane & 16) ? 8 : 0;
    float* base = pnum + (size_t)(h * NC + c) * 64 * 64;
#pragma unroll
    for (int i = 0; i < 4; ++i) {
#pragma unroll
      for (int r = 0; r < 8; ++r) {
        int row = i * 16 + r + rhi;
        base[row * 64 + colp] = acc[i][r];
      }
    }
    if (threadIdx.x < 64)
      pden[(size_t)(h * NC + c) * 64 + threadIdx.x] = den[threadIdx.x];
  }
}

__global__ void attn_combine_kernel(const float* __restrict__ pnum,
                                    const float* __restrict__ pden,
                                    __bf16* __restrict__ ctxb)  // [64 x 1024]
{
  int idx = blockIdx.x * blockDim.x + threadIdx.x;  // b*1024 + h*64 + d
  if (idx >= B_DIM * D_DIM) return;
  int d = idx & 63;
  int h = (idx >> 6) & 15;
  int b = idx >> 10;
  float num = 0.f, den = 0.f;
#pragma unroll
  for (int c = 0; c < NC; ++c) {
    num += pnum[((size_t)(h * NC + c) * 64 + b) * 64 + d];
    den += pden[(size_t)(h * NC + c) * 64 + b];
  }
  ctxb[idx] = (__bf16)(num / den);
}

// vectorized f32 -> bf16 (n % 4 == 0)
__global__ void f2bf4_kernel(const float* __restrict__ s, __bf16* __restrict__ d, int n4) {
  int i = blockIdx.x * blockDim.x + threadIdx.x;
  if (i < n4) {
    float4 f = reinterpret_cast<const float4*>(s)[i];
    union { __bf16 h[4]; uint64_t u; } p;
    p.h[0] = (__bf16)f.x; p.h[1] = (__bf16)f.y;
    p.h[2] = (__bf16)f.z; p.h[3] = (__bf16)f.w;
    reinterpret_cast<uint64_t*>(d)[i] = p.u;
  }
}

__global__ void concat_kernel(const __bf16* __restrict__ a, const __bf16* __restrict__ b,
                              __bf16* __restrict__ o, int rows, int ca, int cb) {
  int i = blockIdx.x * blockDim.x + threadIdx.x;
  int cols = ca + cb;
  if (i >= rows * cols) return;
  int r = i / cols, cc = i % cols;
  o[i] = (cc < ca) ? a[r * ca + cc] : b[r * cb + (cc - ca)];
}

__global__ void gmul_kernel(const __bf16* __restrict__ g, const __bf16* __restrict__ m,
                            __bf16* __restrict__ o, int n) {
  int i = blockIdx.x * blockDim.x + threadIdx.x;
  if (i < n) o[i] = (__bf16)((float)g[i] * (float)m[i]);
}

// ---------------------------------------------------------------------------
extern "C" void kernel_launch(void* const* d_in, const int* in_sizes, int n_in,
                              void* d_out, int out_size, void* d_ws, size_t ws_size,
                              hipStream_t stream)
{
  (void)in_sizes; (void)n_in; (void)out_size; (void)ws_size;

  const float* x    = (const float*)d_in[0];
  const float* mk   = (const float*)d_in[1];
  const float* mv   = (const float*)d_in[2];
  const float* Wq   = (const float*)d_in[3];
  const float* bq   = (const float*)d_in[4];
  const float* Wk   = (const float*)d_in[5];
  const float* bk   = (const float*)d_in[6];
  const float* Wv   = (const float*)d_in[7];
  const float* bvp  = (const float*)d_in[8];
  const float* inW  = (const float*)d_in[9];
  const float* inB  = (const float*)d_in[10];
  const float* outW = (const float*)d_in[11];
  const float* outB = (const float*)d_in[12];
  const float* gW1  = (const float*)d_in[13];
  const float* gb1  = (const float*)d_in[14];
  const float* gW2  = (const float*)d_in[15];
  const float* gb2  = (const float*)d_in[16];
  const float* iW1  = (const float*)d_in[17];
  const float* ib1  = (const float*)d_in[18];
  const float* iW2  = (const float*)d_in[19];
  const float* ib2  = (const float*)d_in[20];
  float* out = (float*)d_out;

  char* ws = (char*)d_ws;
  size_t off = 0;
  auto alloc = [&](size_t bytes) -> void* {
    void* p = ws + off;
    off = (off + bytes + 255) & ~(size_t)255;
    return p;
  };
  const size_t MB1 = (size_t)1024 * 1024;
  // bf16 weight copies (async-copy-friendly)
  __bf16* Wqb  = (__bf16*)alloc(MB1 * 2);
  __bf16* Wkb  = (__bf16*)alloc(MB1 * 2);
  __bf16* Wvb  = (__bf16*)alloc(MB1 * 2);
  __bf16* inWb = (__bf16*)alloc(3 * MB1 * 2);
  __bf16* outWb= (__bf16*)alloc(MB1 * 2);
  __bf16* gW1b = (__bf16*)alloc(2 * MB1 * 2);
  __bf16* gW2b = (__bf16*)alloc(MB1 * 2);
  __bf16* iW1b = (__bf16*)alloc(4 * MB1 * 2);
  __bf16* iW2b = (__bf16*)alloc(2 * MB1 * 2);
  // activations / intermediates
  __bf16* qb      = (__bf16*)alloc((size_t)B_DIM * D_DIM * 2);
  __bf16* qhb     = (__bf16*)alloc((size_t)B_DIM * D_DIM * 2);
  __bf16* kvb     = (__bf16*)alloc((size_t)M_DIM * D_DIM * 2);  // k, then reused for v
  __bf16* khb     = (__bf16*)alloc((size_t)M_DIM * D_DIM * 2);
  __bf16* vhb     = (__bf16*)alloc((size_t)M_DIM * D_DIM * 2);
  float*  pnum    = (float*) alloc((size_t)H_DIM * NC * 64 * 64 * 4);
  float*  pden    = (float*) alloc((size_t)H_DIM * NC * 64 * 4);
  __bf16* ctxb    = (__bf16*)alloc((size_t)B_DIM * D_DIM * 2);
  __bf16* xb      = (__bf16*)alloc((size_t)B_DIM * D_DIM * 2);
  __bf16* msb     = (__bf16*)alloc((size_t)B_DIM * D_DIM * 2);
  __bf16* gateinb = (__bf16*)alloc((size_t)B_DIM * 2 * D_DIM * 2);
  __bf16* g1b     = (__bf16*)alloc((size_t)B_DIM * D_DIM * 2);
  __bf16* gateb   = (__bf16*)alloc((size_t)B_DIM * D_DIM * 2);
  __bf16* gmb     = (__bf16*)alloc((size_t)B_DIM * D_DIM * 2);
  __bf16* combb   = (__bf16*)alloc((size_t)B_DIM * 2 * D_DIM * 2);
  __bf16* i1b     = (__bf16*)alloc((size_t)B_DIM * 2 * D_DIM * 2);

  const dim3 cvt(256);
  auto cv = [&](const float* s, __bf16* d, size_t n) {
    f2bf4_kernel<<<dim3((unsigned)((n / 4 + 255) / 256)), cvt, 0, stream>>>(s, d, (int)(n / 4));
  };
  // one-time bf16 conversions
  cv(x,    xb,    (size_t)B_DIM * D_DIM);
  cv(Wq,   Wqb,   MB1);      cv(Wk, Wkb, MB1);      cv(Wv, Wvb, MB1);
  cv(inW,  inWb,  3 * MB1);  cv(outW, outWb, MB1);
  cv(gW1,  gW1b,  2 * MB1);  cv(gW2, gW2b, MB1);
  cv(iW1,  iW1b,  4 * MB1);  cv(iW2, iW2b, 2 * MB1);

  const dim3 blk(256);

  // q = x @ Wq^T + bq ; qh = q @ Wiq^T + biq
  gemm_bt_kernel<0, true ><<<dim3(8, 1), blk, 0, stream>>>(x,  1024, Wqb,  1024, bq,  qb,  nullptr, 1024, 64, 1024);
  gemm_bt_kernel<0, false><<<dim3(8, 1), blk, 0, stream>>>(qb, 1024, inWb, 1024, inB, qhb, nullptr, 1024, 64, 1024);

  // k = mk @ Wk^T + bk ; kh = k @ Wik^T + bik
  gemm_bt_kernel<0, true ><<<dim3(8, 64), blk, 0, stream>>>(mk,  1024, Wkb, 1024, bk, kvb, nullptr, 1024, 8192, 1024);
  gemm_bt_kernel<0, false><<<dim3(8, 64), blk, 0, stream>>>(kvb, 1024, inWb + MB1, 1024, inB + 1024, khb, nullptr, 1024, 8192, 1024);

  // v = mv @ Wv^T + bv ; vh = v @ Wiv^T + biv   (reuse kvb for v)
  gemm_bt_kernel<0, true ><<<dim3(8, 64), blk, 0, stream>>>(mv,  1024, Wvb, 1024, bvp, kvb, nullptr, 1024, 8192, 1024);
  gemm_bt_kernel<0, false><<<dim3(8, 64), blk, 0, stream>>>(kvb, 1024, inWb + 2 * MB1, 1024, inB + 2048, vhb, nullptr, 1024, 8192, 1024);

  // attention
  attn_partial_kernel<<<dim3(H_DIM, NC), dim3(128), 0, stream>>>(qhb, khb, vhb, pnum, pden);
  attn_combine_kernel<<<dim3(256), dim3(256), 0, stream>>>(pnum, pden, ctxb);

  // memory_signal = ctx @ out_w^T + out_b
  gemm_bt_kernel<0, false><<<dim3(8, 1), blk, 0, stream>>>(ctxb, 1024, outWb, 1024, outB, msb, nullptr, 1024, 64, 1024);

  // gate = sigmoid(relu([x, ms] @ gW1^T + gb1) @ gW2^T + gb2); gm = gate * ms
  concat_kernel<<<dim3(512), dim3(256), 0, stream>>>(xb, msb, gateinb, B_DIM, 1024, 1024);
  gemm_bt_kernel<1, false><<<dim3(8, 1), blk, 0, stream>>>(gateinb, 2048, gW1b, 2048, gb1, g1b,   nullptr, 1024, 64, 2048);
  gemm_bt_kernel<2, false><<<dim3(8, 1), blk, 0, stream>>>(g1b,     1024, gW2b, 1024, gb2, gateb, nullptr, 1024, 64, 1024);
  gmul_kernel<<<dim3(256), dim3(256), 0, stream>>>(gateb, msb, gmb, B_DIM * D_DIM);

  // out = relu([x, gm] @ iW1^T + ib1) @ iW2^T + ib2   (f32 output)
  concat_kernel<<<dim3(512), dim3(256), 0, stream>>>(xb, gmb, combb, B_DIM, 1024, 1024);
  gemm_bt_kernel<1, false><<<dim3(16, 1), blk, 0, stream>>>(combb, 2048, iW1b, 2048, ib1, i1b, nullptr, 2048, 64, 2048);
  gemm_bt_kernel<0, false><<<dim3(8, 1), blk, 0, stream>>>(i1b,  2048, iW2b, 2048, ib2, nullptr, out, 1024, 64, 2048);
}
